// MistralDecoderLayerOA_68788196213058
// MI455X (gfx1250) — matrix operand order repaired and checked by reading, not compiled, not run on hardware
//
#include <hip/hip_runtime.h>

// ---------------------------------------------------------------------------
// MI455X (gfx1250) Mistral decoder layer, BF16 WMMA pipeline with
// TDM (tensor_load_to_lds) A-tile staging + async-to-LDS V staging.
// Shapes: B=1, S=2048, NP=4, HID=4096, H=32, HKV=8, DH=128, FF=14336
// ---------------------------------------------------------------------------

typedef __attribute__((ext_vector_type(16))) __bf16 v16bf;
typedef __attribute__((ext_vector_type(8)))  float  v8f;
typedef __attribute__((ext_vector_type(4)))  unsigned int v4u;
typedef __attribute__((ext_vector_type(8)))  int    v8i;
typedef __attribute__((ext_vector_type(4)))  int    v4i;

union Frag16 { uint4 q[2]; v16bf v; };

__device__ __forceinline__ __bf16 f2bf(float f) { return (__bf16)f; }

__device__ __forceinline__ v8f wmma_bf16(const Frag16& a, const Frag16& b, v8f c) {
    return __builtin_amdgcn_wmma_f32_16x16x32_bf16(false, a.v, false, b.v,
                                                   (short)0, c, false, false);
}

#define S_LEN   2048
#define NP_LEN  4
#define SP_LEN  2052
#define HID_    4096
#define NH      32
#define NKV     8
#define DH_     128
#define FF_     14336

// ---------------------------------------------------------------------------
// TDM: DMA a [rows<=128] x [32] bf16 tile (row stride K elements) into LDS.
// Builds the D# descriptor per CDNA5 ISA 8.3/8.4 (2D tensor, groups 2/3 zero).
// OOB rows (beyond M-m0) read back as zeros.
// 6-arg builtin variant: (v4u g0, v8i g1, v4i g2, v4i g3, v8i g4, i32 cpol)
// ---------------------------------------------------------------------------
__device__ __forceinline__ void tdm_load_a_tile(const __bf16* A, __bf16* sa,
                                                int m0, int k0, int K, int M) {
    unsigned lds  = (unsigned)(size_t)(void*)sa;
    unsigned long long ga = (unsigned long long)(const void*)(A + (size_t)m0 * K + k0);
    int rows = M - m0; if (rows > 128) rows = 128;

    v4u g0;
    g0.x = 1u;                                           // count=1 (valid), load
    g0.y = lds;                                          // lds_addr
    g0.z = (unsigned)(ga & 0xFFFFFFFFu);                 // global_addr[31:0]
    g0.w = (unsigned)((ga >> 32) & 0x01FFFFFFu) | (2u << 30); // addr[56:32] | type=2

    v8i g1;
    g1[0] = (int)(1u << 16);                             // data_size=1 -> 2 bytes
    g1[1] = (int)(32u << 16);                            // tensor_dim0 = 32 (lo16)
    g1[2] = (int)(((unsigned)rows & 0xFFFFu) << 16);     // dim0 hi16 | tensor_dim1 lo16
    g1[3] = (int)(32u << 16);                            // dim1 hi16 | tile_dim0 = 32
    g1[4] = 128;                                         // tile_dim1 = 128
    g1[5] = (int)(unsigned)K;                            // tensor_dim0_stride lo32
    g1[6] = 0;                                           // stride hi | dim1_stride lo
    g1[7] = 0;

    v4i g2 = {0, 0, 0, 0};
    v4i g3 = {0, 0, 0, 0};
    v8i g4 = {0, 0, 0, 0, 0, 0, 0, 0};
    __builtin_amdgcn_tensor_load_to_lds(g0, g1, g2, g3, g4, 0);
}

// ---------------------------------------------------------------------------
// RMSNorm (fp32 in, bf16 out), one block per row, 256 threads
// ---------------------------------------------------------------------------
__global__ __launch_bounds__(256) void k_rmsnorm_bf16(
    const float* __restrict__ X, const float* __restrict__ w,
    __bf16* __restrict__ Y, int ncols)
{
    const int row = blockIdx.x;
    const float* x = X + (size_t)row * ncols;
    float ss = 0.f;
    for (int i = threadIdx.x; i < ncols; i += 256) { float v = x[i]; ss += v * v; }
    #pragma unroll
    for (int off = 16; off >= 1; off >>= 1) ss += __shfl_xor(ss, off, 32);
    __shared__ float red[8];
    __shared__ float s_inv;
    if ((threadIdx.x & 31) == 0) red[threadIdx.x >> 5] = ss;
    __syncthreads();
    if (threadIdx.x == 0) {
        float t = 0.f;
        #pragma unroll
        for (int i = 0; i < 8; i++) t += red[i];
        s_inv = rsqrtf(t / (float)ncols + 1e-5f);
    }
    __syncthreads();
    const float sc = s_inv;
    for (int i = threadIdx.x; i < ncols; i += 256)
        Y[(size_t)row * ncols + i] = f2bf(x[i] * sc * w[i]);
}

// ---------------------------------------------------------------------------
// GEMM: C[M,N] = A[M,K](bf16,row-major) * W[N,K](fp32,row-major)^T
// 128x128x32 tiles, 256 threads = 8 waves, LDS double-buffered.
// A tile: TDM DMA (TENSORcnt). W tile: fp32->bf16 convert, K-major, packed
// b32 stores. mode 0: Cf=acc; 1: Cf=acc+res[min(row,res_rows-1)];
// 2: Cb=bf16(silu(aux)*acc)
// ---------------------------------------------------------------------------
__device__ __forceinline__ void stage_w_tile(const float* __restrict__ W,
                                             __bf16* __restrict__ sw,
                                             int n0, int k0, int K, int tid) {
    const int np = (tid & 63) * 2;          // n-pair base: 0,2,..,126
    const int kk = (tid >> 6) * 8;          // 0,8,16,24
    const float* r0 = W + (size_t)(n0 + np) * K + k0 + kk;
    const float* r1 = r0 + K;
    if (k0 + 32 < K) __builtin_prefetch(r0 + 32, 0, 1);   // global_prefetch_b8
    #pragma unroll
    for (int i = 0; i < 8; i++) {
        union { __bf16 h[2]; unsigned u; } pk;
        pk.h[0] = f2bf(r0[i]);
        pk.h[1] = f2bf(r1[i]);
        *(unsigned*)&sw[(kk + i) * 128 + np] = pk.u;
    }
}

__global__ __launch_bounds__(256) void k_gemm_xwT(
    const __bf16* __restrict__ A, const float* __restrict__ W,
    float* __restrict__ Cf, __bf16* __restrict__ Cb,
    const float* __restrict__ res, const float* __restrict__ aux,
    int M, int N, int K, int mode, int res_rows)
{
    __shared__ __bf16 sA[2][128 * 32];   // row-major [m][k]
    __shared__ __bf16 sW[2][32 * 128];   // K-major  [k][n]

    const int tid  = threadIdx.x;
    const int lane = tid & 31;
    const int wave = tid >> 5;
    const int m0   = blockIdx.x * 128;
    const int n0   = blockIdx.y * 128;
    const int strip = wave * 16;
    const int mlo = lane & 15, mhi = lane >> 4;

    v8f acc[8];
    const v8f vz = {0.f,0.f,0.f,0.f,0.f,0.f,0.f,0.f};
    #pragma unroll
    for (int i = 0; i < 8; i++) acc[i] = vz;

    // ---- prologue: stage tile 0 ----
    if (wave == 0) tdm_load_a_tile(A, &sA[0][0], m0, 0, K, M);
    stage_w_tile(W, &sW[0][0], n0, 0, K, tid);
    if (wave == 0) __builtin_amdgcn_s_wait_tensorcnt(0);
    __syncthreads();

    int buf = 0;
    for (int k0 = 0; k0 < K; k0 += 32, buf ^= 1) {
        const int kn = k0 + 32;
        // ---- stage next tile into the other buffer (overlaps compute) ----
        if (kn < K) {
            if (wave == 0) tdm_load_a_tile(A, &sA[buf ^ 1][0], m0, kn, K, M);
            stage_w_tile(W, &sW[buf ^ 1][0], n0, kn, K, tid);
        }
        // ---- compute on current buffer ----
        Frag16 af;     // ISA 16-bit A layout: two contiguous 8-elem K runs
        {
            const __bf16* base = &sA[buf][(strip + mlo) * 32 + 8 * mhi];
            af.q[0] = *(const uint4*)base;
            af.q[1] = *(const uint4*)(base + 16);
        }
        Frag16 bfr[8]; // lane = output column n, 16 k values in two 8-runs (same k order as A)
        #pragma unroll
        for (int nt = 0; nt < 8; nt++) {
            #pragma unroll
            for (int s = 0; s < 16; s++)
                bfr[nt].v[s] = sW[buf][((s < 8 ? s : s + 8) + 8 * mhi) * 128 + nt * 16 + mlo];
        }
        #pragma unroll
        for (int nt = 0; nt < 8; nt++)
            acc[nt] = wmma_bf16(af, bfr[nt], acc[nt]);

        if (kn < K && wave == 0) __builtin_amdgcn_s_wait_tensorcnt(0);
        __syncthreads();
    }

    // ---- epilogue: C element (m,n): vgpr r -> m = r+8*mhi, n = nt*16+mlo ----
    #pragma unroll
    for (int nt = 0; nt < 8; nt++) {
        #pragma unroll
        for (int r = 0; r < 8; r++) {
            int row = m0 + strip + r + 8 * mhi;
            if (row >= M) continue;
            int col = n0 + nt * 16 + mlo;
            size_t idx = (size_t)row * N + col;
            float v = acc[nt][r];
            if (mode == 0) {
                Cf[idx] = v;
            } else if (mode == 1) {
                int rr = row < res_rows ? row : res_rows - 1;
                Cf[idx] = v + res[(size_t)rr * N + col];
            } else {
                float g = aux[idx];
                float s = g / (1.f + __expf(-g));
                Cb[idx] = f2bf(s * v);
            }
        }
    }
}

// ---------------------------------------------------------------------------
// RoPE + layout shuffle.
//  y in [0,32):   Q head -> Qb[h][s][d], shifted positions; rows s>=S are the
//                 NP prediction rows built from raw_q row p, re-roped.
//  y in [32,40):  K head -> KT[hkv][d][s]   (d-major for attention B frags)
//  y in [40,48):  V head -> Vb[hkv][s][d]   (bf16 copy)
// ---------------------------------------------------------------------------
__global__ __launch_bounds__(64) void k_rope(
    const float* __restrict__ Qr, const float* __restrict__ Kr,
    const float* __restrict__ Vr,
    const int* __restrict__ pos_ids, const int* __restrict__ pos_pred,
    const int* __restrict__ pos_repl,
    __bf16* __restrict__ Qb, __bf16* __restrict__ KT, __bf16* __restrict__ Vb)
{
    const int s = blockIdx.x;      // 0..SP-1
    const int y = blockIdx.y;      // 0..47
    const int d = threadIdx.x;     // 0..63
    // inv_freq = theta^(-2d/128)
    const float inv = __expf(-((float)(2 * d) / 128.f) * 9.210340371976184f);

    if (y < NH) {
        const int h = y;
        const int p = pos_repl[0] - 1;
        const int srow = (s < S_LEN) ? s : p;
        const float* src = Qr + (size_t)srow * (NH * DH_) + h * DH_;
        int pos;
        if (s < S_LEN) pos = (s < S_LEN - 1) ? pos_ids[s + 1] : pos_ids[0];
        else           pos = pos_pred[s - S_LEN];
        const float ang = (float)pos * inv;
        const float c = __cosf(ang), sn = __sinf(ang);
        const float x1 = src[d], x2 = src[d + 64];
        __bf16* dst = Qb + ((size_t)h * SP_LEN + s) * DH_;
        dst[d]      = f2bf(x1 * c - x2 * sn);
        dst[d + 64] = f2bf(x2 * c + x1 * sn);
    } else if (y < NH + NKV) {
        if (s >= S_LEN) return;
        const int hk = y - NH;
        const float* src = Kr + (size_t)s * (NKV * DH_) + hk * DH_;
        const float ang = (float)pos_ids[s] * inv;
        const float c = __cosf(ang), sn = __sinf(ang);
        const float x1 = src[d], x2 = src[d + 64];
        __bf16* dst = KT + (size_t)hk * DH_ * S_LEN;
        dst[(size_t)d * S_LEN + s]        = f2bf(x1 * c - x2 * sn);
        dst[(size_t)(d + 64) * S_LEN + s] = f2bf(x2 * c + x1 * sn);
    } else {
        if (s >= S_LEN) return;
        const int hv = y - NH - NKV;
        const float* src = Vr + (size_t)s * (NKV * DH_) + hv * DH_;
        __bf16* dst = Vb + ((size_t)hv * S_LEN + s) * DH_;
        dst[d]      = f2bf(src[d]);
        dst[d + 64] = f2bf(src[d + 64]);
    }
}

// ---------------------------------------------------------------------------
// Flash attention (GQA, causal; rows >= S use the p = S-1 mask row).
// Block = 128 threads = 4 waves; each wave: 16 query rows. Key blocks of 32.
// V tile staged to LDS with GLOBAL_LOAD_ASYNC_TO_LDS_B128 (ASYNCcnt),
// issued before QK^T so the DMA overlaps the score WMMAs + softmax.
// ---------------------------------------------------------------------------
__global__ __launch_bounds__(128) void k_flash_attn(
    const __bf16* __restrict__ Qb,   // [H][SP][128]
    const __bf16* __restrict__ KT,   // [HKV][128][S]
    const __bf16* __restrict__ Vb,   // [HKV][S][128]
    __bf16* __restrict__ Ob)         // [SP][H*128]
{
    __shared__ __bf16 pl[4][16 * 32];
    __shared__ __bf16 sV[32 * 128];  // [key_local][d]

    const int lane = threadIdx.x & 31;
    const int wave = threadIdx.x >> 5;
    const int h    = blockIdx.x;
    const int q0   = blockIdx.y * 64 + wave * 16;
    const int hkv  = h / (NH / NKV);
    const int mlo  = lane & 15, mhi = lane >> 4;
    const float scale = 0.08838834764831845f;   // 1/sqrt(128)

    // Q fragments: 4 chunks of 32 d-values, kept resident
    Frag16 qf[4];
    {
        int qr = q0 + mlo; if (qr > SP_LEN - 1) qr = SP_LEN - 1;
        const __bf16* qrow = Qb + ((size_t)h * SP_LEN + qr) * DH_;
        #pragma unroll
        for (int dk = 0; dk < 4; dk++) {
            const __bf16* base = qrow + dk * 32 + 8 * mhi;
            qf[dk].q[0] = *(const uint4*)base;
            qf[dk].q[1] = *(const uint4*)(base + 16);
        }
    }

    float row_max[8], row_sum[8];
    v8f oacc[8];
    const v8f vz = {0.f,0.f,0.f,0.f,0.f,0.f,0.f,0.f};
    #pragma unroll
    for (int r = 0; r < 8; r++) { row_max[r] = -3.0e38f; row_sum[r] = 0.f; oacc[r] = vz; }

    // async V staging params: thread -> (key, 32-d segment)
    const int vkey = threadIdx.x >> 2;
    const int vseg = (threadIdx.x & 3) * 32;
    const unsigned vlds = (unsigned)(size_t)(void*)&sV[vkey * DH_ + vseg];

    for (int kb = 0; kb < S_LEN; kb += 32) {
        // ---- kick off async DMA of this key-block's V tile into LDS ----
        __syncthreads();   // prior P*V reads of sV complete
        {
            const __bf16* vsrc = Vb + ((size_t)hkv * S_LEN + kb + vkey) * DH_ + vseg;
            #pragma unroll
            for (int i = 0; i < 4; i++) {
                unsigned la = vlds + i * 16;
                unsigned long long ga = (unsigned long long)(const void*)(vsrc + i * 8);
                asm volatile("global_load_async_to_lds_b128 %0, %1, off"
                             :: "v"(la), "v"(ga) : "memory");
            }
        }

        // ---- scores: two 16x16 tiles over keys kb..kb+31 (overlaps DMA) ----
        v8f s0 = vz, s1 = vz;
        #pragma unroll
        for (int dk = 0; dk < 4; dk++) {
            const __bf16* kcol = KT + ((size_t)hkv * DH_ + dk * 32) * S_LEN + kb + mlo;
            Frag16 b0, b1;
            #pragma unroll
            for (int s = 0; s < 16; s++) {
                const __bf16* kp = kcol + (size_t)((s < 8 ? s : s + 8) + 8 * mhi) * S_LEN;
                b0.v[s] = kp[0];
                b1.v[s] = kp[16];
            }
            s0 = wmma_bf16(qf[dk], b0, s0);
            s1 = wmma_bf16(qf[dk], b1, s1);
        }
        // ---- mask + online softmax, per owned row r ----
        #pragma unroll
        for (int r = 0; r < 8; r++) {
            int qi  = q0 + r + 8 * mhi;
            int lim = qi < S_LEN ? qi : S_LEN - 1;   // ext rows: mask row p
            float v0 = s0[r] * scale, v1 = s1[r] * scale;
            if (kb + mlo > lim)      v0 = -3.0e38f;
            if (kb + 16 + mlo > lim) v1 = -3.0e38f;
            float mx = fmaxf(v0, v1);
            #pragma unroll
            for (int off = 8; off >= 1; off >>= 1) mx = fmaxf(mx, __shfl_xor(mx, off, 32));
            float nm   = fmaxf(row_max[r], mx);
            float corr = __expf(row_max[r] - nm);
            float p0 = __expf(v0 - nm), p1 = __expf(v1 - nm);
            float ps = p0 + p1;
            #pragma unroll
            for (int off = 8; off >= 1; off >>= 1) ps += __shfl_xor(ps, off, 32);
            row_sum[r] = row_sum[r] * corr + ps;
            row_max[r] = nm;
            #pragma unroll
            for (int dt = 0; dt < 8; dt++) oacc[dt][r] *= corr;
            // scatter P (C layout) into wave-private LDS tile [m][key]
            __bf16* pw = &pl[wave][(r + 8 * mhi) * 32];
            pw[mlo]      = f2bf(p0);
            pw[16 + mlo] = f2bf(p1);
        }
        asm volatile("s_wait_dscnt 0" ::: "memory");
        // ---- wait for the V DMA, then P*V from LDS ----
        asm volatile("s_wait_asynccnt 0" ::: "memory");
        __syncthreads();

        Frag16 pf;
        {
            const __bf16* base = &pl[wave][mlo * 32 + 8 * mhi];
            pf.q[0] = *(const uint4*)base;
            pf.q[1] = *(const uint4*)(base + 16);
        }
        #pragma unroll
        for (int dt = 0; dt < 8; dt++) {
            Frag16 bv;
            #pragma unroll
            for (int s = 0; s < 16; s++)
                bv.v[s] = sV[((s < 8 ? s : s + 8) + 8 * mhi) * DH_ + dt * 16 + mlo];
            oacc[dt] = wmma_bf16(pf, bv, oacc[dt]);
        }
    }

    // ---- normalize + store: Ob[row][h*128 + d] ----
    #pragma unroll
    for (int r = 0; r < 8; r++) {
        int row = q0 + r + 8 * mhi;
        if (row >= SP_LEN) continue;
        float inv = 1.f / row_sum[r];
        #pragma unroll
        for (int dt = 0; dt < 8; dt++)
            Ob[(size_t)row * (NH * DH_) + h * DH_ + dt * 16 + mlo] =
                f2bf(oacc[dt][r] * inv);
    }
}

// ---------------------------------------------------------------------------
// Host launcher
// ---------------------------------------------------------------------------
extern "C" void kernel_launch(void* const* d_in, const int* in_sizes, int n_in,
                              void* d_out, int out_size, void* d_ws, size_t ws_size,
                              hipStream_t stream) {
    const float* hidden   = (const float*)d_in[0];
    // d_in[1] attention_mask: causal structure computed analytically
    const int*   pos_ids  = (const int*)d_in[2];
    const int*   pos_repl = (const int*)d_in[3];
    const int*   pos_pred = (const int*)d_in[4];
    const float* Wq = (const float*)d_in[5];
    const float* Wk = (const float*)d_in[6];
    const float* Wv = (const float*)d_in[7];
    const float* Wo = (const float*)d_in[8];
    const float* Wg = (const float*)d_in[9];
    const float* Wu = (const float*)d_in[10];
    const float* Wd = (const float*)d_in[11];
    const float* ln1 = (const float*)d_in[12];
    const float* ln2 = (const float*)d_in[13];

    char* ws = (char*)d_ws;
    size_t off = 0;
    auto take = [&](size_t bytes) -> void* {
        void* p = ws + off;
        off += (bytes + 255) & ~(size_t)255;
        return p;
    };
    __bf16* xbf  = (__bf16*)take((size_t)S_LEN * HID_ * 2);
    float*  qraw = (float*) take((size_t)S_LEN * HID_ * 4);
    float*  kraw = (float*) take((size_t)S_LEN * NKV * DH_ * 4);
    float*  vraw = (float*) take((size_t)S_LEN * NKV * DH_ * 4);
    __bf16* qb   = (__bf16*)take((size_t)NH * SP_LEN * DH_ * 2);
    __bf16* kT   = (__bf16*)take((size_t)NKV * DH_ * S_LEN * 2);
    __bf16* vb   = (__bf16*)take((size_t)NKV * S_LEN * DH_ * 2);
    __bf16* aout = (__bf16*)take((size_t)SP_LEN * HID_ * 2);
    float*  hbuf = (float*) take((size_t)SP_LEN * HID_ * 4);
    __bf16* ybf  = (__bf16*)take((size_t)SP_LEN * HID_ * 2);
    float*  gbuf = (float*) take((size_t)SP_LEN * FF_ * 4);
    __bf16* abf  = (__bf16*)take((size_t)SP_LEN * FF_ * 2);
    (void)ws_size; (void)in_sizes; (void)n_in; (void)out_size;

    // 1) RMSNorm(hidden) -> x (bf16)
    k_rmsnorm_bf16<<<S_LEN, 256, 0, stream>>>(hidden, ln1, xbf, HID_);

    // 2) QKV projections
    k_gemm_xwT<<<dim3(S_LEN/128, HID_/128), 256, 0, stream>>>(
        xbf, Wq, qraw, nullptr, nullptr, nullptr, S_LEN, HID_, HID_, 0, 0);
    k_gemm_xwT<<<dim3(S_LEN/128, (NKV*DH_)/128), 256, 0, stream>>>(
        xbf, Wk, kraw, nullptr, nullptr, nullptr, S_LEN, NKV*DH_, HID_, 0, 0);
    k_gemm_xwT<<<dim3(S_LEN/128, (NKV*DH_)/128), 256, 0, stream>>>(
        xbf, Wv, vraw, nullptr, nullptr, nullptr, S_LEN, NKV*DH_, HID_, 0, 0);

    // 3) RoPE + attention-friendly layouts (incl. NP prediction query rows)
    k_rope<<<dim3(SP_LEN, NH + 2*NKV), 64, 0, stream>>>(
        qraw, kraw, vraw, pos_ids, pos_pred, pos_repl, qb, kT, vb);

    // 4) Flash attention over all SP = S+NP query rows
    k_flash_attn<<<dim3(NH, (SP_LEN + 63) / 64), 128, 0, stream>>>(qb, kT, vb, aout);

    // 5) O-projection + residual (ext rows reuse hidden row p = S-1)
    k_gemm_xwT<<<dim3((SP_LEN+127)/128, HID_/128), 256, 0, stream>>>(
        aout, Wo, hbuf, nullptr, hidden, nullptr, SP_LEN, HID_, HID_, 1, S_LEN);

    // 6) RMSNorm(h) -> y (bf16)
    k_rmsnorm_bf16<<<SP_LEN, 256, 0, stream>>>(hbuf, ln2, ybf, HID_);

    // 7) MLP: gate, then up fused with silu(gate)*up -> bf16
    k_gemm_xwT<<<dim3((SP_LEN+127)/128, FF_/128), 256, 0, stream>>>(
        ybf, Wg, gbuf, nullptr, nullptr, nullptr, SP_LEN, FF_, HID_, 0, 0);
    k_gemm_xwT<<<dim3((SP_LEN+127)/128, FF_/128), 256, 0, stream>>>(
        ybf, Wu, nullptr, abf, nullptr, gbuf, SP_LEN, FF_, HID_, 2, 0);

    // 8) down-projection + residual h -> final output (fp32)
    k_gemm_xwT<<<dim3((SP_LEN+127)/128, HID_/128), 256, 0, stream>>>(
        abf, Wd, (float*)d_out, nullptr, hbuf, nullptr, SP_LEN, HID_, FF_, 1, SP_LEN);
}
